// DecoderGroupedQueryHeadAttentionAlibi_80736795231130
// MI455X (gfx1250) — compile-verified, matrix operand order checked by reading, not run on hardware
//
#include <hip/hip_runtime.h>
#include <math.h>

// ---------------------------------------------------------------------------
// Types for CDNA5 WMMA (gfx1250, wave32)
// ---------------------------------------------------------------------------
typedef __attribute__((ext_vector_type(16))) __bf16       v16bf;
typedef __attribute__((ext_vector_type(8)))  float        v8f;
typedef __attribute__((ext_vector_type(4))) unsigned int  u32x4;

union Frag16 {           // 16 bf16 = one A/B fragment (32 bytes)
    u32x4 u[2];
    v16bf v;
};
union VStage {           // one staged 16-byte V segment (8 bf16)
    u32x4 u;
    unsigned short h[8];
};

__device__ __forceinline__ v8f wmma_bf16(v16bf a, v16bf b, v8f c) {
    return __builtin_amdgcn_wmma_f32_16x16x32_bf16(
        /*neg_a=*/false, a, /*neg_b=*/false, b,
        /*c_mod=*/(short)0, c, /*reuse_a=*/false, /*reuse_b=*/false);
}

__device__ __forceinline__ unsigned short f2bf(float f) {
    unsigned u = __builtin_bit_cast(unsigned, f);
    u += 0x7FFFu + ((u >> 16) & 1u);   // round-to-nearest-even
    return (unsigned short)(u >> 16);
}

// A-fragment (16x32, MxK) or B-fragment (col-per-lane) load from row-major
// bf16 (global or LDS). Per ISA 7.12.2: lane L holds row (L&15); lo half
// covers K kb..kb+7, hi half covers K kb+16..kb+23, kb = 8*(L>=16).
__device__ __forceinline__ v16bf load_frag16(const unsigned short* base,
                                             int stride, int rowBase,
                                             int kOff, int lane) {
    int r  = rowBase + (lane & 15);
    int kb = kOff + ((lane >> 4) << 3);
    const unsigned short* p = base + (size_t)r * stride + kb;
    Frag16 f;
    f.u[0] = *(const u32x4*)(p);
    f.u[1] = *(const u32x4*)(p + 16);
    return f.v;
}

// ---- CDNA5 async global->LDS (tracked by ASYNCcnt) -------------------------
// VGLOBAL async encoding: VDST = LDS byte-offset VGPR, ADDR = 64-bit global.
#define ASYNC_G2L_B128(ldsoff, gaddr)                                          \
    asm volatile("global_load_async_to_lds_b128 %0, %1, off"                   \
                 :: "v"(ldsoff), "v"(gaddr) : "memory")

__device__ __forceinline__ void s_wait_asynccnt_0() {
    asm volatile("s_wait_asynccnt 0x0" ::: "memory");
}
__device__ __forceinline__ void s_wait_asynccnt_1() {
    asm volatile("s_wait_asynccnt 0x1" ::: "memory");
}
__device__ __forceinline__ void s_wait_asynccnt_3() {
    asm volatile("s_wait_asynccnt 0x3" ::: "memory");
}

// generic shared pointer -> LDS byte offset (aperture lives in the high bits)
__device__ __forceinline__ unsigned lds_off(const void* p) {
    return (unsigned)(size_t)p;
}

// Wave-local LDS ordering: all prior ds ops complete before later ones issue.
__device__ __forceinline__ void wave_sync_lds() {
    __builtin_amdgcn_wave_barrier();
    asm volatile("s_wait_dscnt 0x0" ::: "memory");
    __builtin_amdgcn_wave_barrier();
}

// ---------------------------------------------------------------------------
// Problem constants
// ---------------------------------------------------------------------------
constexpr int BATCH = 4;
constexpr int SEQ   = 2048;
constexpr int NH    = 16;
constexpr int NKV   = 4;
constexpr int HD    = 64;
constexpr int EMB   = 1024;   // NH*HD
constexpr int KVE   = 512;    // 2*NKV*HD

// ---------------------------------------------------------------------------
// Stage 0: fp32 -> bf16 conversion
// ---------------------------------------------------------------------------
__global__ __launch_bounds__(256) void conv_f32_bf16(const float* __restrict__ in,
                                                     unsigned short* __restrict__ out,
                                                     int n) {
    int i = blockIdx.x * blockDim.x + threadIdx.x;
    int stride = gridDim.x * blockDim.x;
    for (; i < n; i += stride) out[i] = f2bf(in[i]);
}

// ---------------------------------------------------------------------------
// Stage 1/3: GEMM  out[m,n] = sum_k A[m,k] * W[n,k]  (+bias), bf16 WMMA.
// Block = 8 waves; block tile 128(M) x 64(N); K-steps of 32 double-buffered
// through LDS via global_load_async_to_lds_b128 (ASYNCcnt pipeline).
// ---------------------------------------------------------------------------
__global__ __launch_bounds__(256) void gemm_bf16_wmma(
    const unsigned short* __restrict__ A,   // [M,K] bf16
    const unsigned short* __restrict__ W,   // [N,K] bf16
    const float* __restrict__ bias,         // [N] or nullptr
    float* __restrict__ outF,               // fp32 out (or nullptr)
    unsigned short* __restrict__ outB,      // bf16 out (or nullptr)
    int M, int N, int K)
{
    __shared__ alignas(16) unsigned short Abuf[2][128 * 32];   // 2 x 8 KB
    __shared__ alignas(16) unsigned short Wbuf[2][64 * 32];    // 2 x 4 KB

    int tid   = threadIdx.x;
    int lane  = tid & 31;
    int w     = tid >> 5;
    int mTile = blockIdx.y * 128;
    int nTile = blockIdx.x * 64;

    // 16-byte staging segments: A has 512 (2/thread), W has 256 (1/thread).
    int aRow0 = tid >> 2,        aCol0 = (tid & 3) * 8;
    int aRow1 = (tid >> 2) + 64, aCol1 = aCol0;
    int wRow  = tid >> 2,        wCol  = aCol0;

    const unsigned short* aG0 = A + (size_t)(mTile + aRow0) * K + aCol0;
    const unsigned short* aG1 = A + (size_t)(mTile + aRow1) * K + aCol1;
    const unsigned short* wG  = W + (size_t)(nTile + wRow)  * K + wCol;

    unsigned aL0[2], aL1[2], wL[2];
    #pragma unroll
    for (int bb = 0; bb < 2; ++bb) {
        aL0[bb] = lds_off(&Abuf[bb][aRow0 * 32 + aCol0]);
        aL1[bb] = lds_off(&Abuf[bb][aRow1 * 32 + aCol1]);
        wL[bb]  = lds_off(&Wbuf[bb][wRow  * 32 + wCol ]);
    }

    auto issue = [&](int bb, int kk) {
        ASYNC_G2L_B128(aL0[bb], (unsigned long long)(aG0 + kk));
        ASYNC_G2L_B128(aL1[bb], (unsigned long long)(aG1 + kk));
        ASYNC_G2L_B128(wL[bb],  (unsigned long long)(wG  + kk));
    };

    issue(0, 0);

    v8f acc[4] = {};
    const int nsteps = K >> 5;
    for (int s = 0; s < nsteps; ++s) {
        int bb = s & 1;
        if (s + 1 < nsteps) {
            issue(bb ^ 1, (s + 1) << 5);   // prefetch next tile into other buf
            s_wait_asynccnt_3();           // current tile's 3 ops are done
        } else {
            s_wait_asynccnt_0();
        }
        __syncthreads();                   // all waves' stages landed

        v16bf a = load_frag16(&Abuf[bb][0], 32, w * 16, 0, lane);
        #pragma unroll
        for (int g = 0; g < 4; ++g) {
            v16bf bf = load_frag16(&Wbuf[bb][0], 32, g * 16, 0, lane);
            acc[g] = wmma_bf16(a, bf, acc[g]);
        }
        __syncthreads();                   // done reading before next overwrite
    }

    int colL = lane & 15, hi = lane >> 4;
    #pragma unroll
    for (int g = 0; g < 4; ++g) {
        #pragma unroll
        for (int r = 0; r < 8; ++r) {
            int row = mTile + w * 16 + r + 8 * hi;
            int col = nTile + g * 16 + colL;
            float v = acc[g][r];
            if (bias) v += bias[col];
            if (outF) outF[(size_t)row * N + col] = v;
            else      outB[(size_t)row * N + col] = f2bf(v);
        }
    }
}

// ---------------------------------------------------------------------------
// Stage 2: fused flash attention with lower-triangular ALiBi bias (no causal
// mask: bias = 0 above diagonal, softmax over the full row — faithful to ref).
// Grid: (SEQ/128, NH, BATCH); 8 waves/block, each wave owns 16 query rows.
// Block-cooperative chunk loop: K chunks async-staged to LDS (double buffer),
// V chunks transposed into LDS so P·V fragments come from ds_load_b128.
// ---------------------------------------------------------------------------
__global__ __launch_bounds__(256) void flash_attn_alibi(
    const unsigned short* __restrict__ qB,   // [B*T, 1024] bf16
    const unsigned short* __restrict__ kvB,  // [B*T, 512]  bf16 (k | v)
    unsigned short* __restrict__ aOut)       // [B*T, 1024] bf16
{
    int tid  = threadIdx.x;
    int lane = tid & 31;
    int w    = tid >> 5;
    int qt   = blockIdx.x;
    int h    = blockIdx.y;
    int b    = blockIdx.z;
    int kvh  = h & (NKV - 1);
    int qW   = qt * 128 + w * 16;            // this wave's query-row base

    __shared__ alignas(16) unsigned short Kbuf[2][32 * 64];   // 2 x 4 KB
    __shared__ alignas(16) unsigned short Vtb[2][64 * 32];    // 2 x 4 KB (Vt[d][kk])
    __shared__ alignas(16) float          Sbuf[8][16][32];
    __shared__ alignas(16) unsigned short Pbuf[8][16][32];
    __shared__ float AlphaBuf[8][16];
    __shared__ float Lbuf[8][16];

    const unsigned short* qBase = qB  + (size_t)b * SEQ * EMB + h * HD;
    const unsigned short* kBase = kvB + (size_t)b * SEQ * KVE + kvh * HD;
    const unsigned short* vBase = kvB + (size_t)b * SEQ * KVE + NKV * HD + kvh * HD;

    // Staging assignment: 256 threads x one 16-byte segment per chunk.
    int sRow = tid >> 3;          // key row within chunk: 0..31
    int sCol = (tid & 7) * 8;     // d offset: 0..56
    const unsigned short* kG = kBase + (size_t)sRow * KVE + sCol;
    const unsigned short* vG = vBase + (size_t)sRow * KVE + sCol;
    unsigned kL[2] = { lds_off(&Kbuf[0][sRow * 64 + sCol]),
                       lds_off(&Kbuf[1][sRow * 64 + sCol]) };

    auto issueK = [&](int bb, int c0) {
        ASYNC_G2L_B128(kL[bb], (unsigned long long)(kG + (size_t)c0 * KVE));
    };

    // Q fragments (d 0..31 and 32..63) held in registers for the whole loop.
    v16bf aq0 = load_frag16(qBase + 0,  EMB, qW, 0, lane);
    v16bf aq1 = load_frag16(qBase + 32, EMB, qW, 0, lane);

    v8f   o[4] = {};
    float mRun = -1.0e30f, lRun = 0.0f;
    const float slope = exp2f(-0.5f * (float)(h + 1)) * 0.125f; // alibi * D^-0.5

    int colL = lane & 15, hi = lane >> 4;
    int kb = hi << 3;

    // Prologue: chunk 0 in flight (K async to LDS, V to registers).
    VStage vreg, vnext;
    issueK(0, 0);
    vreg.u = *(const u32x4*)(vG);

    const int NCH = SEQ / 32;
    for (int s = 0; s < NCH; ++s) {
        int bb = s & 1;
        int c0 = s * 32;
        if (s + 1 < NCH) issueK(bb ^ 1, c0 + 32);   // next K chunk, async

        // Transpose-store this chunk's V into Vt[d][kk].
        #pragma unroll
        for (int e = 0; e < 8; ++e)
            Vtb[bb][(sCol + e) * 32 + sRow] = vreg.h[e];

        if (s + 1 < NCH) {
            vnext.u = *(const u32x4*)(vG + (size_t)(c0 + 32) * KVE);
            s_wait_asynccnt_1();                    // current K chunk landed
        } else {
            s_wait_asynccnt_0();
        }
        __syncthreads();                            // K/Vt ready block-wide

        // ---- S = Q·K^T for 32 keys (two 16-key tiles, K-dim = 64) ----
        #pragma unroll
        for (int t = 0; t < 2; ++t) {
            v16bf bk0 = load_frag16(&Kbuf[bb][0],      64, t * 16, 0, lane);
            v16bf bk1 = load_frag16(&Kbuf[bb][0] + 32, 64, t * 16, 0, lane);
            v8f sAcc = {};
            sAcc = wmma_bf16(aq0, bk0, sAcc);
            sAcc = wmma_bf16(aq1, bk1, sAcc);
            #pragma unroll
            for (int r = 0; r < 8; ++r)
                Sbuf[w][r + 8 * hi][t * 16 + colL] = sAcc[r];
        }
        wave_sync_lds();

        // ---- scale + ALiBi + online softmax (one row per lane 0..15) ----
        if (lane < 16) {
            int rr = lane;
            int iG = qW + rr;
            float mold = mRun, mnew = mold;
            #pragma unroll
            for (int j = 0; j < 32; ++j) {
                int jG  = c0 + j;
                float sv = Sbuf[w][rr][j] * 0.125f;
                if (jG <= iG) sv += slope * (float)(jG - iG);
                Sbuf[w][rr][j] = sv;
                mnew = fmaxf(mnew, sv);
            }
            float lsum = 0.0f;
            #pragma unroll
            for (int j = 0; j < 32; ++j) {
                float p = __expf(Sbuf[w][rr][j] - mnew);
                lsum += p;
                Pbuf[w][rr][j] = f2bf(p);
            }
            float alpha = __expf(mold - mnew);
            lRun = lRun * alpha + lsum;
            mRun = mnew;
            AlphaBuf[w][rr] = alpha;
        }
        wave_sync_lds();

        // ---- rescale accumulators ----
        #pragma unroll
        for (int r = 0; r < 8; ++r) {
            float al = AlphaBuf[w][r + 8 * hi];
            #pragma unroll
            for (int g = 0; g < 4; ++g) o[g][r] = o[g][r] * al;
        }

        // ---- O += P·V (P: 16x32 A-frag; V: B-frags from transposed LDS) ----
        v16bf pf = load_frag16(&Pbuf[w][0][0], 32, 0, 0, lane);
        #pragma unroll
        for (int g = 0; g < 4; ++g) {
            v16bf fv = load_frag16(&Vtb[bb][0], 32, g * 16, 0, lane);
            o[g] = wmma_bf16(pf, fv, o[g]);
        }
        __syncthreads();                            // done with Kbuf/Vtb[bb]
        if (s + 1 < NCH) vreg = vnext;
    }

    // ---- finalize: divide by row sums, write bf16 [B*T, 1024] ----
    if (lane < 16) Lbuf[w][lane] = lRun;
    wave_sync_lds();
    #pragma unroll
    for (int r = 0; r < 8; ++r) {
        float inv = 1.0f / Lbuf[w][r + 8 * hi];
        int row = qW + r + 8 * hi;
        #pragma unroll
        for (int g = 0; g < 4; ++g)
            aOut[((size_t)b * SEQ + row) * EMB + h * HD + g * 16 + colL] =
                f2bf(o[g][r] * inv);
    }
}

// ---------------------------------------------------------------------------
// Host orchestration
// ---------------------------------------------------------------------------
extern "C" void kernel_launch(void* const* d_in, const int* in_sizes, int n_in,
                              void* d_out, int out_size, void* d_ws, size_t ws_size,
                              hipStream_t stream) {
    (void)in_sizes; (void)n_in; (void)out_size; (void)ws_size;

    const float* x     = (const float*)d_in[0];   // [4,2048,1024]
    const float* Wq    = (const float*)d_in[1];   // [1024,1024]
    const float* Wkv   = (const float*)d_in[2];   // [512,1024]
    const float* Wproj = (const float*)d_in[3];   // [1024,1024]
    const float* bproj = (const float*)d_in[4];   // [1024]

    const int M = BATCH * SEQ;                    // 8192

    // Workspace carve-up (bf16 buffers), 256-B aligned.
    char* ws = (char*)d_ws;
    size_t off = 0;
    auto carve = [&](size_t bytes) {
        void* p = ws + off;
        off += (bytes + 255) & ~(size_t)255;
        return p;
    };
    unsigned short* xb  = (unsigned short*)carve((size_t)M * EMB * 2);   // 16 MB
    unsigned short* qb  = (unsigned short*)carve((size_t)M * EMB * 2);   // 16 MB
    unsigned short* kvb = (unsigned short*)carve((size_t)M * KVE * 2);   //  8 MB
    unsigned short* ab  = (unsigned short*)carve((size_t)M * EMB * 2);   // 16 MB
    unsigned short* wqb = (unsigned short*)carve((size_t)EMB * EMB * 2); //  2 MB
    unsigned short* wkb = (unsigned short*)carve((size_t)KVE * EMB * 2); //  1 MB
    unsigned short* wpb = (unsigned short*)carve((size_t)EMB * EMB * 2); //  2 MB

    // Stage 0: convert to bf16
    conv_f32_bf16<<<dim3(4096), dim3(256), 0, stream>>>(x,     xb,  M * EMB);
    conv_f32_bf16<<<dim3(2048), dim3(256), 0, stream>>>(Wq,    wqb, EMB * EMB);
    conv_f32_bf16<<<dim3(1024), dim3(256), 0, stream>>>(Wkv,   wkb, KVE * EMB);
    conv_f32_bf16<<<dim3(2048), dim3(256), 0, stream>>>(Wproj, wpb, EMB * EMB);

    // Stage 1: q = x·Wq^T  and  kv = x·Wkv^T  (bf16 out)
    gemm_bf16_wmma<<<dim3(EMB / 64, M / 128), dim3(256), 0, stream>>>(
        xb, wqb, nullptr, nullptr, qb, M, EMB, EMB);
    gemm_bf16_wmma<<<dim3(KVE / 64, M / 128), dim3(256), 0, stream>>>(
        xb, wkb, nullptr, nullptr, kvb, M, KVE, EMB);

    // Stage 2: fused flash attention with ALiBi
    flash_attn_alibi<<<dim3(SEQ / 128, NH, BATCH), dim3(256), 0, stream>>>(
        qb, kvb, ab);

    // Stage 3: out = attn·Wproj^T + bproj (fp32 out)
    gemm_bf16_wmma<<<dim3(EMB / 64, M / 128), dim3(256), 0, stream>>>(
        ab, wpb, bproj, (float*)d_out, nullptr, M, EMB, EMB);
}
